// SpecialSelfAttention_40647570489844
// MI455X (gfx1250) — compile-verified, matrix operand order checked by reading, not run on hardware
//
#include <hip/hip_runtime.h>

typedef __attribute__((ext_vector_type(16))) __bf16 v16bf;
typedef __attribute__((ext_vector_type(8)))  __bf16 v8bf;
typedef __attribute__((ext_vector_type(8)))  float  v8f;

#define NFULL   2048
#define CDIM    320
#define NHEAD   5
#define NAB     20                 // B*H
#define ABSTRIDE (NFULL * 64)      // 131072 elements per attention batch

// ---------------------------------------------------------------------------
// CDNA5 async global->LDS copy (ASYNCcnt path), guarded so compile never breaks.
// Probe-derived prototype: param0 = AS(1) pointer to int __vector_size__(16).
// ---------------------------------------------------------------------------
#if defined(__gfx1250__) && __has_builtin(__builtin_amdgcn_global_load_async_to_lds_b128)
#define USE_ASYNC_LDS 1
#else
#define USE_ASYNC_LDS 0
#endif

typedef int v4i32 __attribute__((vector_size(16)));
typedef __attribute__((address_space(1))) v4i32 as1_v4i32;
typedef __attribute__((address_space(3))) v4i32 as3_v4i32;

__device__ __forceinline__ void async_copy_b128(const __bf16* g, __bf16* l) {
#if USE_ASYNC_LDS
  __builtin_amdgcn_global_load_async_to_lds_b128(
      (as1_v4i32*)(unsigned long long)g,
      (as3_v4i32*)(unsigned long long)l, 0, 0);
#else
  *reinterpret_cast<v8bf*>(l) = *reinterpret_cast<const v8bf*>(g);
#endif
}

__device__ __forceinline__ void wait_async_copies() {
#if USE_ASYNC_LDS
#if __has_builtin(__builtin_amdgcn_s_wait_asynccnt)
  __builtin_amdgcn_s_wait_asynccnt(0);
#else
  asm volatile("s_wait_asynccnt 0" ::: "memory");
#endif
#endif
}

__device__ __forceinline__ v8f wmma_bf16(v16bf a, v16bf b, v8f c) {
  // D = A(16x32 bf16) * B(32x16 bf16) + C(16x16 f32)
  return __builtin_amdgcn_wmma_f32_16x16x32_bf16(false, a, false, b, (short)0, c,
                                                 false, false);
}

// 16-bit operand fragment for one lane from a 32-element contiguous K-run.
// elems 0..7 <- K = half*8+0..7 ; elems 8..15 <- K = 16+half*8+0..7
__device__ __forceinline__ v16bf load_frag_bf(const __bf16* p, int half) {
  v8bf lo = *reinterpret_cast<const v8bf*>(p + half * 8);
  v8bf hi = *reinterpret_cast<const v8bf*>(p + 16 + half * 8);
  v16bf f;
#pragma unroll
  for (int i = 0; i < 8; ++i) { f[i] = lo[i]; f[8 + i] = hi[i]; }
  return f;
}

__device__ __forceinline__ v16bf load_frag_f32(const float* p, int half) {
  v16bf f;
#pragma unroll
  for (int i = 0; i < 8; ++i) f[i] = (__bf16)p[half * 8 + i];
#pragma unroll
  for (int i = 0; i < 8; ++i) f[8 + i] = (__bf16)p[16 + half * 8 + i];
  return f;
}

// ---------------------------------------------------------------------------
// Kernel 1: QKV projection. Y = X @ W, head-split, edge/scale folded in.
//   Qs[ab][n][d] = (x@Wq)[n, h*64+d] * (1/8) * edge[ab%4][n]
//   Ks[ab][n][d] = (x@Wk)[n, h*64+d] * edge[ab%4][n]
//   Vt[ab][d][n] = (x@Wv)[n, h*64+d]          (transposed for PV B-operand)
// ---------------------------------------------------------------------------
__global__ __launch_bounds__(256)
void qkv_proj_kernel(const float* __restrict__ x, const float* __restrict__ edge,
                     const float* __restrict__ Wq, const float* __restrict__ Wk,
                     const float* __restrict__ Wv,
                     __bf16* __restrict__ Qs, __bf16* __restrict__ Ks,
                     __bf16* __restrict__ Vt) {
  __shared__ __bf16 wt[64 * 32];   // weight tile, transposed: [col][k]
  const int which = blockIdx.z;
  const float* W = (which == 0) ? Wq : ((which == 1) ? Wk : Wv);
  const int head = blockIdx.y;
  const int tid  = threadIdx.x;
  const int lane = tid & 31, wid = tid >> 5;
  const int m = lane & 15, half = lane >> 4;
  const int rowbase = blockIdx.x * 128 + wid * 16;

  const v8f vzero = {0, 0, 0, 0, 0, 0, 0, 0};
  v8f c[4] = {vzero, vzero, vzero, vzero};

  for (int kb = 0; kb < CDIM; kb += 32) {
    __syncthreads();
    for (int e = tid; e < 2048; e += 256) {
      int k = e >> 6, col = e & 63;
      wt[col * 32 + k] = (__bf16)W[(kb + k) * CDIM + head * 64 + col];
    }
    __syncthreads();
    const float* xp = x + (rowbase + m) * CDIM + kb;
    v16bf a = load_frag_f32(xp, half);
#pragma unroll
    for (int t = 0; t < 4; ++t) {
      v16bf b = load_frag_bf(&wt[(t * 16 + m) * 32], half);
      c[t] = wmma_bf16(a, b, c[t]);
    }
  }

  const int bidx  = rowbase >> 11;
  const int nbase = rowbase & (NFULL - 1);
  const int ab    = bidx * NHEAD + head;
  const int eb    = ab & 3;                  // edge index = ab % B
  const long abOff = (long)ab * ABSTRIDE;

#pragma unroll
  for (int r = 0; r < 8; ++r) {
    const int n = nbase + r + half * 8;
    float f = 1.0f;
    if (which == 0)      f = 0.125f * edge[eb * NFULL + n];   // scale = D^-0.5
    else if (which == 1) f = edge[eb * NFULL + n];
#pragma unroll
    for (int t = 0; t < 4; ++t) {
      const int d = t * 16 + m;
      const float val = c[t][r] * f;
      if (which == 0)      Qs[abOff + (long)n * 64 + d]    = (__bf16)val;
      else if (which == 1) Ks[abOff + (long)n * 64 + d]    = (__bf16)val;
      else                 Vt[abOff + (long)d * NFULL + n] = (__bf16)val;
    }
  }
}

// ---------------------------------------------------------------------------
// Kernel 2: flash attention, block-cooperative.
// 8 waves/block share one ab; K/V tiles (32 cols) double-buffered in LDS via
// async global->LDS copies; each wave owns 16 query rows x D=64.
// grid 320 blocks = 20 ab * 16 qblocks(128 rows).
// ---------------------------------------------------------------------------
__global__ __launch_bounds__(256)
void flash_attn_kernel(const __bf16* __restrict__ Qs, const __bf16* __restrict__ Ks,
                       const __bf16* __restrict__ Vt, __bf16* __restrict__ Obf) {
  __shared__ __bf16 kt[2][32 * 64];     // K tile [j][d], double buffered (2x4KB)
  __shared__ __bf16 vt[2][64 * 32];     // V^T tile [d][j], double buffered (2x4KB)
  __shared__ __bf16 pbuf[8 * 512];      // per-wave 16x32 P tile (8KB)

  const int tid  = threadIdx.x;
  const int lane = tid & 31, wid = tid >> 5;
  const int m = lane & 15, half = lane >> 4;
  const int task    = blockIdx.x * 8 + wid;
  const int ab      = task >> 7;                 // all 8 waves: same ab
  const int rowbase = (task & 127) << 4;

  const __bf16* Qb = Qs + (long)ab * ABSTRIDE;
  const __bf16* Kb = Ks + (long)ab * ABSTRIDE;
  const __bf16* Vb = Vt + (long)ab * ABSTRIDE;
  __bf16* pb = pbuf + wid * 512;

  // cooperative tile fetch: 256 threads x one B128 each per tile
  const int kj = tid >> 3, kd = (tid & 7) * 8;   // K: j row, d chunk
  const int vd = tid >> 2, vj = (tid & 3) * 8;   // V: d row, j chunk

  const __bf16* qrow = Qb + (rowbase + m) * 64;
  const v16bf aq0 = load_frag_bf(qrow, half);        // d = 0..31
  const v16bf aq1 = load_frag_bf(qrow + 32, half);   // d = 32..63

  const v8f vzero = {0, 0, 0, 0, 0, 0, 0, 0};
  v8f o[4] = {vzero, vzero, vzero, vzero};
  float mr[8], lr[8];
#pragma unroll
  for (int r = 0; r < 8; ++r) { mr[r] = -1.0e30f; lr[r] = 0.0f; }

  // prologue: fetch tile 0 into buffer 0
  async_copy_b128(Kb + kj * 64 + kd, &kt[0][kj * 64 + kd]);
  async_copy_b128(Vb + vd * NFULL + vj, &vt[0][vd * 32 + vj]);

  int buf = 0;
  for (int jb = 0; jb < NFULL; jb += 32) {
    wait_async_copies();       // this wave's portion landed in LDS
    __syncthreads();           // all waves' portions landed; prev compute done

    if (jb + 32 < NFULL) {     // overlap next tile fetch with current compute
      const int jn = jb + 32;
      async_copy_b128(Kb + (jn + kj) * 64 + kd, &kt[buf ^ 1][kj * 64 + kd]);
      async_copy_b128(Vb + vd * NFULL + jn + vj, &vt[buf ^ 1][vd * 32 + vj]);
      __builtin_prefetch(Kb + (jb + 64 + kj) * 64 + kd, 0, 3);
    }

    // ---- S = Q K^T on current tile (columns jb..jb+31) ----
    const __bf16* kc0 = &kt[buf][m * 64];          // cols j = jb+0..15
    const __bf16* kc1 = &kt[buf][(16 + m) * 64];   // cols j = jb+16..31
    v8f s0 = vzero, s1 = vzero;
    s0 = wmma_bf16(aq0, load_frag_bf(kc0, half), s0);
    s0 = wmma_bf16(aq1, load_frag_bf(kc0 + 32, half), s0);
    s1 = wmma_bf16(aq0, load_frag_bf(kc1, half), s1);
    s1 = wmma_bf16(aq1, load_frag_bf(kc1 + 32, half), s1);

    // ---- online softmax over the 32 new columns ----
    float corr[8];
#pragma unroll
    for (int r = 0; r < 8; ++r) {
      float tm = fmaxf(s0[r], s1[r]);              // butterfly within 16-lane half
      tm = fmaxf(tm, __shfl_xor(tm, 1, 32));
      tm = fmaxf(tm, __shfl_xor(tm, 2, 32));
      tm = fmaxf(tm, __shfl_xor(tm, 4, 32));
      tm = fmaxf(tm, __shfl_xor(tm, 8, 32));
      const float mn = fmaxf(mr[r], tm);
      corr[r] = __expf(mr[r] - mn);
      mr[r] = mn;
      const float p0 = __expf(s0[r] - mn);
      const float p1 = __expf(s1[r] - mn);
      s0[r] = p0; s1[r] = p1;
      float rs = p0 + p1;
      rs += __shfl_xor(rs, 1, 32);
      rs += __shfl_xor(rs, 2, 32);
      rs += __shfl_xor(rs, 4, 32);
      rs += __shfl_xor(rs, 8, 32);
      lr[r] = lr[r] * corr[r] + rs;
    }
#pragma unroll
    for (int r = 0; r < 8; ++r) {
      o[0][r] *= corr[r]; o[1][r] *= corr[r];
      o[2][r] *= corr[r]; o[3][r] *= corr[r];
    }

    // ---- accumulator layout -> A layout via per-wave LDS tile ----
#pragma unroll
    for (int r = 0; r < 8; ++r) {
      const int row = r + half * 8;
      pb[row * 32 + m]      = (__bf16)s0[r];
      pb[row * 32 + 16 + m] = (__bf16)s1[r];
    }
    asm volatile("s_wait_dscnt 0" ::: "memory");
    const v16bf ap = load_frag_bf(pb + m * 32, half);

    // ---- O += P V ----
#pragma unroll
    for (int t = 0; t < 4; ++t)
      o[t] = wmma_bf16(ap, load_frag_bf(&vt[buf][(t * 16 + m) * 32], half), o[t]);

    __syncthreads();           // everyone done reading tile[buf] before reuse
    buf ^= 1;
  }

  __bf16* ob = Obf + (long)ab * ABSTRIDE;
#pragma unroll
  for (int r = 0; r < 8; ++r) {
    const float inv = 1.0f / lr[r];
    const int n = rowbase + r + half * 8;
#pragma unroll
    for (int t = 0; t < 4; ++t)
      ob[(long)n * 64 + t * 16 + m] = (__bf16)(o[t][r] * inv);
  }
}

// ---------------------------------------------------------------------------
// Kernel 3: merge heads + output projection: out = O_merged @ Wo + bo (fp32)
// ---------------------------------------------------------------------------
__global__ __launch_bounds__(256)
void out_proj_kernel(const __bf16* __restrict__ Obf, const float* __restrict__ Wo,
                     const float* __restrict__ bo, float* __restrict__ out) {
  __shared__ __bf16 wt[64 * 32];
  const int tid  = threadIdx.x;
  const int lane = tid & 31, wid = tid >> 5;
  const int m = lane & 15, half = lane >> 4;
  const int rowbase = blockIdx.x * 128 + wid * 16;
  const int colbase = blockIdx.y * 64;

  const v8f vzero = {0, 0, 0, 0, 0, 0, 0, 0};
  v8f c[4] = {vzero, vzero, vzero, vzero};

  const int gr = rowbase + m;
  const int b  = gr >> 11;
  const int n  = gr & (NFULL - 1);

  for (int kb = 0; kb < CDIM; kb += 32) {
    __syncthreads();
    for (int e = tid; e < 2048; e += 256) {
      int k = e >> 6, col = e & 63;
      wt[col * 32 + k] = (__bf16)Wo[(kb + k) * CDIM + colbase + col];
    }
    __syncthreads();
    // gather merged-head A fragment: k = h*64 + d  -> Obf[(b*5+h)][n][d]
    const int ks0 = kb + half * 8;
    const int ks1 = kb + 16 + half * 8;
    const __bf16* p0 = Obf + ((long)(b * NHEAD + (ks0 >> 6)) * NFULL + n) * 64 + (ks0 & 63);
    const __bf16* p1 = Obf + ((long)(b * NHEAD + (ks1 >> 6)) * NFULL + n) * 64 + (ks1 & 63);
    v8bf lo = *reinterpret_cast<const v8bf*>(p0);
    v8bf hi = *reinterpret_cast<const v8bf*>(p1);
    v16bf a;
#pragma unroll
    for (int i = 0; i < 8; ++i) { a[i] = lo[i]; a[8 + i] = hi[i]; }
#pragma unroll
    for (int t = 0; t < 4; ++t) {
      v16bf bb = load_frag_bf(&wt[(t * 16 + m) * 32], half);
      c[t] = wmma_bf16(a, bb, c[t]);
    }
  }

#pragma unroll
  for (int r = 0; r < 8; ++r) {
    const int orow = rowbase + r + half * 8;
#pragma unroll
    for (int t = 0; t < 4; ++t) {
      const int col = colbase + t * 16 + m;
      out[(long)orow * CDIM + col] = c[t][r] + bo[col];
    }
  }
}

// ---------------------------------------------------------------------------
extern "C" void kernel_launch(void* const* d_in, const int* in_sizes, int n_in,
                              void* d_out, int out_size, void* d_ws, size_t ws_size,
                              hipStream_t stream) {
  const float* x    = (const float*)d_in[0];
  const float* edge = (const float*)d_in[1];
  const float* Wq   = (const float*)d_in[2];
  const float* Wk   = (const float*)d_in[3];
  const float* Wv   = (const float*)d_in[4];
  const float* Wo   = (const float*)d_in[5];
  const float* bo   = (const float*)d_in[6];
  float* out = (float*)d_out;

  __bf16* ws = (__bf16*)d_ws;
  const long seg = (long)NAB * ABSTRIDE;     // 2,621,440 bf16 elements (5.24 MB)
  __bf16* Qs  = ws;                          // edge*scale-folded Q  (ab, n, d)
  __bf16* Ks  = ws + seg;                    // edge-folded K        (ab, n, d)
  __bf16* Vt  = ws + 2 * seg;                // V transposed         (ab, d, n)
  __bf16* Obf = ws + 3 * seg;                // attention output     (ab, n, d)

  qkv_proj_kernel<<<dim3(64, 5, 3), 256, 0, stream>>>(x, edge, Wq, Wk, Wv, Qs, Ks, Vt);
  flash_attn_kernel<<<dim3(320), 256, 0, stream>>>(Qs, Ks, Vt, Obf);
  out_proj_kernel<<<dim3(64, 5), 256, 0, stream>>>(Obf, Wo, bo, out);
}
